// MMDLossSqrt_75711683493952
// MI455X (gfx1250) — matrix operand order repaired and checked by reading, not compiled, not run on hardware
//
#include <hip/hip_runtime.h>
#include <hip/hip_bf16.h>
#include <math.h>

// ---------------------------------------------------------------------------
// MMD loss (Gaussian kernel, sigma=32) fully fused on gfx1250:
//   - bf16x3 split-GEMM via v_wmma_f32_16x16x32_bf16 (near-fp32 accuracy)
//   - Gram matrices never materialized: exp + reduce fused in epilogue
//   - symmetric (xx, yy) computed on upper-triangle tiles only
//   - fast path: one-shot bf16 hi/lo precompute in workspace +
//     global_load_async_to_lds_b128 staging (ASYNCcnt)
//   - fallback path (small ws): fp32 load + in-register split (round-1 kernel)
//   - deterministic (no float atomics): per-block partials + fixed-tree reduce
// ---------------------------------------------------------------------------

typedef __attribute__((ext_vector_type(16))) __bf16          v16bf;
typedef __attribute__((ext_vector_type(8)))  __bf16          v8bf;
typedef __attribute__((ext_vector_type(8)))  float           v8f;
typedef __attribute__((ext_vector_type(4)))  float           v4f;
typedef __attribute__((ext_vector_type(4)))  unsigned short  us4;

#define TILE        128
#define KSTEP       32
#define LDS_STRIDE  40   // 32 bf16 + 8 pad (ushorts) -> conflict-free b128 reads
#define SIGMA2_INV  (1.0f / (32.0f * 32.0f))

__device__ __forceinline__ unsigned short f32_to_bf16_rne(float f) {
    unsigned int u = __float_as_uint(f);
    unsigned int r = u + 0x7FFFu + ((u >> 16) & 1u);   // round to nearest even
    return (unsigned short)(r >> 16);
}
__device__ __forceinline__ float bf16_bits_to_f32(unsigned short h) {
    return __uint_as_float(((unsigned int)h) << 16);
}

// Async copy of 16 bytes / lane: global -> LDS, tracked by ASYNCcnt.
__device__ __forceinline__ void async_copy_b128(const unsigned short* gsrc,
                                                unsigned short* ldst) {
    unsigned lds_off = (unsigned)(uintptr_t)ldst;   // low 32 bits = LDS offset
    asm volatile("global_load_async_to_lds_b128 %0, %1, off"
                 :: "v"(lds_off), "v"(gsrc) : "memory");
}
__device__ __forceinline__ void wait_asynccnt0() {
    asm volatile("s_wait_asynccnt 0" ::: "memory");
}

// A-fragment (16-bit, 16x32): lane holds row M=lane%16; K chunks
//   lanes 0-15 : K [0..8) and [16..24)   lanes 16-31: K [8..16) and [24..32)
__device__ __forceinline__ v16bf load_a_frag(const unsigned short* rowp, int hi16) {
    const __bf16* p = (const __bf16*)rowp;
    v8bf c0 = *(const v8bf*)(p + hi16 * 8);
    v8bf c1 = *(const v8bf*)(p + hi16 * 8 + 16);
    return __builtin_shufflevector(c0, c1, 0,1,2,3,4,5,6,7,8,9,10,11,12,13,14,15);
}
// B-fragment (16-bit, 32x16): lane holds column N=lane%16;
//   lanes 0-15 : K [0..8) and [16..24)   lanes 16-31: K [8..16) and [24..32)
__device__ __forceinline__ v16bf load_b_frag(const unsigned short* rowp, int hi16) {
    const __bf16* p = (const __bf16*)rowp;
    v8bf c0 = *(const v8bf*)(p + hi16 * 8);
    v8bf c1 = *(const v8bf*)(p + hi16 * 8 + 16);
    return __builtin_shufflevector(c0, c1, 0,1,2,3,4,5,6,7,8,9,10,11,12,13,14,15);
}

// ---------------------------------------------------------------------------
__global__ void mmd_row_norms(const float* __restrict__ x,
                              const float* __restrict__ y,
                              float* __restrict__ norms, int N, int D) {
    int row = blockIdx.x;
    const float* src = (row < N) ? (x + (size_t)row * D)
                                 : (y + (size_t)(row - N) * D);
    float s = 0.f;
    for (int c = threadIdx.x; c < D; c += blockDim.x) {
        float v = src[c];
        s += v * v;
    }
    for (int off = 16; off > 0; off >>= 1) s += __shfl_down(s, off, 32);
    __shared__ float wsum[8];
    int lane = threadIdx.x & 31, w = threadIdx.x >> 5;
    if (lane == 0) wsum[w] = s;
    __syncthreads();
    if (threadIdx.x == 0) {
        float t = 0.f;
        for (int i = 0; i < 8; ++i) t += wsum[i];
        norms[row] = t;
    }
}

// ---------------------------------------------------------------------------
// One-shot fp32 -> (bf16 hi, bf16 lo) split into workspace.
__global__ void mmd_split_bf16(const float* __restrict__ src,
                               unsigned short* __restrict__ hi,
                               unsigned short* __restrict__ lo, int n4) {
    int i = blockIdx.x * blockDim.x + threadIdx.x;
    if (i < n4) {
        v4f v = ((const v4f*)src)[i];
        us4 h, l;
        #pragma unroll
        for (int e = 0; e < 4; ++e) {
            unsigned short hb = f32_to_bf16_rne(v[e]);
            h[e] = hb;
            l[e] = f32_to_bf16_rne(v[e] - bf16_bits_to_f32(hb));
        }
        ((us4*)hi)[i] = h;
        ((us4*)lo)[i] = l;
    }
}

// ---------------------------------------------------------------------------
// Shared epilogue + reduction (fused exp and deterministic block partial).
__device__ __forceinline__ void mmd_epilogue(v8f acc[2][4],
                                             const float* __restrict__ normA,
                                             const float* __restrict__ normB,
                                             float* __restrict__ partial,
                                             int rowBase, int colBase,
                                             int waveRow, int waveCol,
                                             int lr, int hi16, int lane, int wave,
                                             int t, int ti, int tj, int symmetric) {
    float lsum = 0.f;
    #pragma unroll
    for (int tc = 0; tc < 4; ++tc) {
        float nb = normB[colBase + waveCol * 64 + tc * 16 + lr];
        #pragma unroll
        for (int tr = 0; tr < 2; ++tr) {
            int mBase = rowBase + waveRow * 32 + tr * 16 + hi16 * 8;
            v8f c = acc[tr][tc];
            #pragma unroll
            for (int r = 0; r < 8; ++r) {
                float d2 = normA[mBase + r] + nb - 2.0f * c[r];
                lsum += __expf(-d2 * SIGMA2_INV);
            }
        }
    }
    if (symmetric && ti != tj) lsum *= 2.0f;   // off-diagonal tiles count twice

    for (int off = 16; off > 0; off >>= 1) lsum += __shfl_down(lsum, off, 32);
    __shared__ float wsum[8];
    __syncthreads();
    if (lane == 0) wsum[wave] = lsum;
    __syncthreads();
    if (t == 0) {
        float s = 0.f;
        for (int i = 0; i < 8; ++i) s += wsum[i];
        partial[blockIdx.x] = s;
    }
}

__device__ __forceinline__ void decode_tile(int tilesPerDim, int symmetric,
                                            int* ti, int* tj) {
    if (symmetric) {
        int rem = blockIdx.x; int i = 0;
        while (rem >= tilesPerDim - i) { rem -= tilesPerDim - i; ++i; }
        *ti = i; *tj = i + rem;
    } else {
        *ti = blockIdx.x / tilesPerDim;
        *tj = blockIdx.x % tilesPerDim;
    }
}

// ---------------------------------------------------------------------------
// Fast path: operands are precomputed bf16 hi/lo arrays; staging is pure
// async DMA (global_load_async_to_lds_b128), compute is bf16x3 WMMA.
__global__ __launch_bounds__(256)
void mmd_gram_exp_sum_bf16(const unsigned short* __restrict__ AHi,
                           const unsigned short* __restrict__ ALo,
                           const unsigned short* __restrict__ BHi,
                           const unsigned short* __restrict__ BLo,
                           const float* __restrict__ normA,
                           const float* __restrict__ normB,
                           float* __restrict__ partial,
                           int D, int tilesPerDim, int symmetric) {
    __shared__ unsigned short aHi[TILE * LDS_STRIDE];
    __shared__ unsigned short aLo[TILE * LDS_STRIDE];
    __shared__ unsigned short bHi[TILE * LDS_STRIDE];
    __shared__ unsigned short bLo[TILE * LDS_STRIDE];

    int ti, tj;
    decode_tile(tilesPerDim, symmetric, &ti, &tj);
    const int rowBase = ti * TILE;
    const int colBase = tj * TILE;

    const int t    = threadIdx.x;
    const int lane = t & 31;
    const int wave = t >> 5;
    const int waveRow = wave >> 1;
    const int waveCol = wave & 1;
    const int lr   = lane & 15;
    const int hi16 = (lane >> 4) & 1;

    v8f zero;
    #pragma unroll
    for (int i = 0; i < 8; ++i) zero[i] = 0.f;
    v8f acc[2][4];
    #pragma unroll
    for (int a = 0; a < 2; ++a)
        #pragma unroll
        for (int b = 0; b < 4; ++b) acc[a][b] = zero;

    const int kSteps = D / KSTEP;
    #pragma unroll 1
    for (int ks = 0; ks < kSteps; ++ks) {
        const int k0 = ks * KSTEP;
        __syncthreads();   // all waves done reading LDS from previous step
        // stage 4x (128 x 32) bf16 slabs: 2 chunks of 16B per thread per slab
        #pragma unroll
        for (int i = 0; i < 2; ++i) {
            int cid = i * 256 + t;            // 0..511 (4 chunks per row)
            int r   = cid >> 2;
            int c8  = (cid & 3) * 8;
            size_t gA = (size_t)(rowBase + r) * D + k0 + c8;
            size_t gB = (size_t)(colBase + r) * D + k0 + c8;
            int l = r * LDS_STRIDE + c8;
            async_copy_b128(AHi + gA, &aHi[l]);
            async_copy_b128(ALo + gA, &aLo[l]);
            async_copy_b128(BHi + gB, &bHi[l]);
            async_copy_b128(BLo + gB, &bLo[l]);
        }
        wait_asynccnt0();   // this wave's async writes landed in LDS
        __syncthreads();    // -> everyone's writes visible

        v16bf aFH[2], aFL[2], bFH[4], bFL[4];
        #pragma unroll
        for (int tr = 0; tr < 2; ++tr) {
            int row = waveRow * 32 + tr * 16 + lr;
            aFH[tr] = load_a_frag(&aHi[row * LDS_STRIDE], hi16);
            aFL[tr] = load_a_frag(&aLo[row * LDS_STRIDE], hi16);
        }
        #pragma unroll
        for (int tc = 0; tc < 4; ++tc) {
            int row = waveCol * 64 + tc * 16 + lr;
            bFH[tc] = load_b_frag(&bHi[row * LDS_STRIDE], hi16);
            bFL[tc] = load_b_frag(&bLo[row * LDS_STRIDE], hi16);
        }
        #pragma unroll
        for (int tr = 0; tr < 2; ++tr)
            #pragma unroll
            for (int tc = 0; tc < 4; ++tc) {
                v8f c = acc[tr][tc];
                c = __builtin_amdgcn_wmma_f32_16x16x32_bf16(false, aFH[tr], false, bFH[tc], (short)0, c, false, false);
                c = __builtin_amdgcn_wmma_f32_16x16x32_bf16(false, aFH[tr], false, bFL[tc], (short)0, c, false, false);
                c = __builtin_amdgcn_wmma_f32_16x16x32_bf16(false, aFL[tr], false, bFH[tc], (short)0, c, false, false);
                acc[tr][tc] = c;
            }
    }

    mmd_epilogue(acc, normA, normB, partial, rowBase, colBase,
                 waveRow, waveCol, lr, hi16, lane, wave, t, ti, tj, symmetric);
}

// ---------------------------------------------------------------------------
// Fallback path (tiny workspace): fp32 loads + in-register bf16 split.
__global__ __launch_bounds__(256)
void mmd_gram_exp_sum_f32(const float* __restrict__ A, const float* __restrict__ B,
                          const float* __restrict__ normA, const float* __restrict__ normB,
                          float* __restrict__ partial, int D, int tilesPerDim, int symmetric) {
    __shared__ unsigned short aHi[TILE * LDS_STRIDE];
    __shared__ unsigned short aLo[TILE * LDS_STRIDE];
    __shared__ unsigned short bHi[TILE * LDS_STRIDE];
    __shared__ unsigned short bLo[TILE * LDS_STRIDE];

    int ti, tj;
    decode_tile(tilesPerDim, symmetric, &ti, &tj);
    const int rowBase = ti * TILE;
    const int colBase = tj * TILE;

    const int t    = threadIdx.x;
    const int lane = t & 31;
    const int wave = t >> 5;
    const int waveRow = wave >> 1;
    const int waveCol = wave & 1;
    const int lr   = lane & 15;
    const int hi16 = (lane >> 4) & 1;

    v8f zero;
    #pragma unroll
    for (int i = 0; i < 8; ++i) zero[i] = 0.f;
    v8f acc[2][4];
    #pragma unroll
    for (int a = 0; a < 2; ++a)
        #pragma unroll
        for (int b = 0; b < 4; ++b) acc[a][b] = zero;

    const int kSteps = D / KSTEP;
    #pragma unroll 1
    for (int ks = 0; ks < kSteps; ++ks) {
        const int k0 = ks * KSTEP;
        __syncthreads();
        #pragma unroll
        for (int i = 0; i < 4; ++i) {
            int idx = i * 1024 + t * 4;
            int r = idx >> 5;
            int c = idx & 31;
            v4f va = *(const v4f*)(A + (size_t)(rowBase + r) * D + k0 + c);
            v4f vb = *(const v4f*)(B + (size_t)(colBase + r) * D + k0 + c);
            us4 ah, al, bh, bl;
            #pragma unroll
            for (int e = 0; e < 4; ++e) {
                unsigned short h = f32_to_bf16_rne(va[e]);
                ah[e] = h;  al[e] = f32_to_bf16_rne(va[e] - bf16_bits_to_f32(h));
                h = f32_to_bf16_rne(vb[e]);
                bh[e] = h;  bl[e] = f32_to_bf16_rne(vb[e] - bf16_bits_to_f32(h));
            }
            *(us4*)&aHi[r * LDS_STRIDE + c] = ah;
            *(us4*)&aLo[r * LDS_STRIDE + c] = al;
            *(us4*)&bHi[r * LDS_STRIDE + c] = bh;
            *(us4*)&bLo[r * LDS_STRIDE + c] = bl;
        }
        __syncthreads();

        v16bf aFH[2], aFL[2], bFH[4], bFL[4];
        #pragma unroll
        for (int tr = 0; tr < 2; ++tr) {
            int row = waveRow * 32 + tr * 16 + lr;
            aFH[tr] = load_a_frag(&aHi[row * LDS_STRIDE], hi16);
            aFL[tr] = load_a_frag(&aLo[row * LDS_STRIDE], hi16);
        }
        #pragma unroll
        for (int tc = 0; tc < 4; ++tc) {
            int row = waveCol * 64 + tc * 16 + lr;
            bFH[tc] = load_b_frag(&bHi[row * LDS_STRIDE], hi16);
            bFL[tc] = load_b_frag(&bLo[row * LDS_STRIDE], hi16);
        }
        #pragma unroll
        for (int tr = 0; tr < 2; ++tr)
            #pragma unroll
            for (int tc = 0; tc < 4; ++tc) {
                v8f c = acc[tr][tc];
                c = __builtin_amdgcn_wmma_f32_16x16x32_bf16(false, aFH[tr], false, bFH[tc], (short)0, c, false, false);
                c = __builtin_amdgcn_wmma_f32_16x16x32_bf16(false, aFH[tr], false, bFL[tc], (short)0, c, false, false);
                c = __builtin_amdgcn_wmma_f32_16x16x32_bf16(false, aFL[tr], false, bFH[tc], (short)0, c, false, false);
                acc[tr][tc] = c;
            }
    }

    mmd_epilogue(acc, normA, normB, partial, rowBase, colBase,
                 waveRow, waveCol, lr, hi16, lane, wave, t, ti, tj, symmetric);
}

// ---------------------------------------------------------------------------
__global__ void mmd_finalize(const float* __restrict__ pXX, int nXX,
                             const float* __restrict__ pYY, int nYY,
                             const float* __restrict__ pXY, int nXY,
                             float* __restrict__ out, int N, int M) {
    __shared__ double sh[256];
    const float* ptrs[3] = {pXX, pYY, pXY};
    const int    cnts[3] = {nXX, nYY, nXY};
    double sums[3];
    for (int s = 0; s < 3; ++s) {
        double acc = 0.0;
        for (int i = threadIdx.x; i < cnts[s]; i += (int)blockDim.x)
            acc += (double)ptrs[s][i];
        sh[threadIdx.x] = acc;
        __syncthreads();
        for (int off = 128; off > 0; off >>= 1) {
            if ((int)threadIdx.x < off) sh[threadIdx.x] += sh[threadIdx.x + off];
            __syncthreads();
        }
        sums[s] = sh[0];
        __syncthreads();
    }
    if (threadIdx.x == 0) {
        double dN = (double)N, dM = (double)M;
        double mmd = (sums[0] - dN) / (dN * (dN - 1.0))
                   + (sums[1] - dM) / (dM * (dM - 1.0))
                   - 2.0 * sums[2] / (dN * dM);
        out[0] = (float)sqrt(fabs(mmd));
    }
}

// ---------------------------------------------------------------------------
extern "C" void kernel_launch(void* const* d_in, const int* in_sizes, int n_in,
                              void* d_out, int out_size, void* d_ws, size_t ws_size,
                              hipStream_t stream) {
    (void)n_in; (void)out_size;
    const float* x = (const float*)d_in[0];
    const float* y = (const float*)d_in[1];
    const int D = 512;
    const int N = in_sizes[0] / D;   // 8192
    const int M = in_sizes[1] / D;   // 8192

    const int tilesN = N / TILE, tilesM = M / TILE;
    const int nXX = tilesN * (tilesN + 1) / 2;
    const int nYY = tilesM * (tilesM + 1) / 2;
    const int nXY = tilesN * tilesM;

    float* normX = (float*)d_ws;          // N
    float* normY = normX + N;             // M
    float* pXX   = normY + M;             // nXX
    float* pYY   = pXX + nXX;             // nYY
    float* pXY   = pYY + nYY;             // nXY

    const size_t elemsX = (size_t)N * D, elemsY = (size_t)M * D;
    size_t offF = (size_t)(N + M) + nXX + nYY + nXY;
    offF = (offF + 3) & ~(size_t)3;       // 16B-align the bf16 region
    const size_t need = offF * 4 + (elemsX + elemsY) * 2 * 2;

    mmd_row_norms<<<N + M, 256, 0, stream>>>(x, y, normX, N, D);

    if (ws_size >= need) {
        // Fast path: precompute bf16 hi/lo once; async-DMA staging in GEMM.
        unsigned short* xHi = (unsigned short*)((float*)d_ws + offF);
        unsigned short* xLo = xHi + elemsX;
        unsigned short* yHi = xLo + elemsX;
        unsigned short* yLo = yHi + elemsY;
        const int n4x = (int)(elemsX / 4), n4y = (int)(elemsY / 4);
        mmd_split_bf16<<<(n4x + 255) / 256, 256, 0, stream>>>(x, xHi, xLo, n4x);
        mmd_split_bf16<<<(n4y + 255) / 256, 256, 0, stream>>>(y, yHi, yLo, n4y);

        mmd_gram_exp_sum_bf16<<<nXX, 256, 0, stream>>>(xHi, xLo, xHi, xLo,
                                                       normX, normX, pXX, D, tilesN, 1);
        mmd_gram_exp_sum_bf16<<<nYY, 256, 0, stream>>>(yHi, yLo, yHi, yLo,
                                                       normY, normY, pYY, D, tilesM, 1);
        mmd_gram_exp_sum_bf16<<<nXY, 256, 0, stream>>>(xHi, xLo, yHi, yLo,
                                                       normX, normY, pXY, D, tilesM, 0);
    } else {
        // Fallback: fp32 loads with in-register split (no big workspace).
        mmd_gram_exp_sum_f32<<<nXX, 256, 0, stream>>>(x, x, normX, normX, pXX, D, tilesN, 1);
        mmd_gram_exp_sum_f32<<<nYY, 256, 0, stream>>>(y, y, normY, normY, pYY, D, tilesM, 1);
        mmd_gram_exp_sum_f32<<<nXY, 256, 0, stream>>>(x, y, normX, normY, pXY, D, tilesM, 0);
    }

    mmd_finalize<<<1, 256, 0, stream>>>(pXX, nXX, pYY, nYY, pXY, nXY,
                                        (float*)d_out, N, M);
}
